// TypeDP_19207093747722
// MI455X (gfx1250) — matrix operand order repaired and checked by reading, compiled, not run on hardware
//
#include <hip/hip_runtime.h>
#include <hip/hip_bf16.h>

typedef _Float16 half_t;
typedef __attribute__((ext_vector_type(16))) _Float16 v16h;
typedef __attribute__((ext_vector_type(8)))  _Float16 v8h;
typedef __attribute__((ext_vector_type(8)))  float    v8f;

#define NTYPES_ 2
#define MNT_ 200
#define M1_ 100
#define M2_ 16
#define FH_ 240
#define BB_ 4
#define NAT_ 1024
#define NATOMS_ (BB_ * NAT_)   // 4096
#define SMEM_A 133632

union V16U { v16h v; v8h h[2]; };

static __device__ __forceinline__ v8f wmma32(v16h a, v16h b, v8f c) {
  // v_wmma_f32_16x16x32_f16  D = A(16x32) * B(32x16) + C
  return __builtin_amdgcn_wmma_f32_16x16x32_f16(false, a, false, b, (short)0, c, false, false);
}

// CDNA5 native transcendental tanh (V_TANH_F32, TRANS32 pipe).
static __device__ __forceinline__ float fast_tanh(float x) {
#if __has_builtin(__builtin_amdgcn_tanhf)
  return __builtin_amdgcn_tanhf(x);
#else
  float r;
  asm volatile("v_tanh_f32 %0, %1\n\tv_nop" : "=v"(r) : "v"(x));
  return r;
#endif
}

// A fragment (16 rows x 32 K), row-major with `stride` halves (mult of 8).
// ISA 7.12.2: lanes 0-15: v0..3=K(0..7), v4..7=K(16..23); lanes 16-31: +8.
static __device__ __forceinline__ v16h load_a(const half_t* base, int stride, int k0, int lane) {
  int row = lane & 15, kh = lane >> 4;
  const half_t* p = base + row * stride + k0 + kh * 8;
  V16U u;
  u.h[0] = *(const v8h*)(p);
  u.h[1] = *(const v8h*)(p + 16);
  return u.v;
}

// B fragment (32 K x 16 N), column-major storage b[n*kstride + k].
// lanes 0-15: N=lane, K=0..7,16..23; lanes 16-31: N=lane-16, K=8..15,24..31.
static __device__ __forceinline__ v16h load_b(const half_t* base, int kstride, int n0, int k0, int lane) {
  int n = n0 + (lane & 15), kq = lane >> 4;
  const half_t* p = base + n * kstride + k0 + kq * 8;
  V16U u;
  u.h[0] = *(const v8h*)(p);
  u.h[1] = *(const v8h*)(p + 16);
  return u.v;
}

// ---------------------------------------------------------------------------
// Kernel A: per-atom embedding MLP (WMMA) + xyz GEMM (WMMA) + DR -> scratch f16
// One block (256 threads, 8 waves) per atom. Dynamic LDS = 133632 B (<320KB/WGP).
// ---------------------------------------------------------------------------
__global__ __launch_bounds__(256) void embed_kernel(
    const float* __restrict__ ImageDR, const int* __restrict__ typemap,
    const float* __restrict__ davg, const float* __restrict__ dstd,
    const float* __restrict__ tv,  const float* __restrict__ tW0, const float* __restrict__ tb0,
    const float* __restrict__ eW0, const float* __restrict__ eb0,
    const float* __restrict__ eW1, const float* __restrict__ eb1,
    const float* __restrict__ eW2, const float* __restrict__ eb2,
    half_t* __restrict__ DRg)
{
  extern __shared__ char smem[];
  half_t* X0  = (half_t*)(smem);            // 208 x 32 (K pad 9->32)
  half_t* H1  = (half_t*)(smem + 13312);    // 208 x 32 (25->32)
  half_t* H2  = (half_t*)(smem + 26624);    // 208 x 64 (50->64)
  half_t* G   = (half_t*)(smem + 53248);    // col-major: G[m*224 + j], 112 x 224
  half_t* RiT = (half_t*)(smem + 103424);   // 16 x 224 (rows 0..3 = Ri^T)
  half_t* W0  = (half_t*)(smem + 110592);   // col-major n*32+k, 32n x 32k
  half_t* W1  = (half_t*)(smem + 112640);   // 64n x 32k
  half_t* W2  = (half_t*)(smem + 116736);   // 112n x 64k
  float*  BS  = (float*)(smem + 131072);    // b0@0(25) b1@25(50) b2@75(100)
  float*  TFE = (float*)(smem + 131776);    // 2 x 8 type features
  float*  XYZ = (float*)(smem + 131840);    // 4 x 112 f32

  const int tid  = threadIdx.x;
  const int lane = tid & 31;
  const int wave = tid >> 5;
  const int a    = blockIdx.x;           // global atom (b*NAT + n)
  const int natom = a & (NAT_ - 1);
  const int t_a  = typemap[natom];

  // ---- phase 0: zero NaN-sensitive pads, type features, weights+biases ----
  for (int i = tid; i < 16 * 224; i += 256) RiT[i] = (half_t)0.f;               // rows>=4, cols>=200 stay 0
  for (int i = tid; i < 8 * 32;   i += 256) X0[200 * 32 + i] = (half_t)0.f;     // rows 200..207
  for (int i = tid; i < 112 * 24; i += 256) {                                   // G K-pad rows j=200..223
    int m = i / 24, j = 200 + (i % 24);
    G[m * 224 + j] = (half_t)0.f;
  }
  if (tid < 16) { // type-embedding net: tanh(tv@tW0+tb0) + [tv,tv]
    int tt = tid >> 3, f = tid & 7;
    float s = tb0[f];
    for (int p = 0; p < 4; ++p) s += tv[tt * 4 + p] * tW0[p * 8 + f];
    TFE[tid] = fast_tanh(s) + tv[tt * 4 + (f & 3)];
  }
  for (int i = tid; i < 32 * 32;  i += 256) { int n = i >> 5, k = i & 31; W0[i] = (half_t)((k < 9  && n < 25)  ? eW0[k * 25  + n] : 0.f); }
  for (int i = tid; i < 64 * 32;  i += 256) { int n = i >> 5, k = i & 31; W1[i] = (half_t)((k < 25 && n < 50)  ? eW1[k * 50  + n] : 0.f); }
  for (int i = tid; i < 112 * 64; i += 256) { int n = i >> 6, k = i & 63; W2[i] = (half_t)((k < 50 && n < 100) ? eW2[k * 100 + n] : 0.f); }
  for (int i = tid; i < 25;  i += 256) BS[i]      = eb0[i];
  for (int i = tid; i < 50;  i += 256) BS[25 + i] = eb1[i];
  for (int i = tid; i < 100; i += 256) BS[75 + i] = eb2[i];
  __syncthreads();

  // ---- phase 1: smooth-cutoff descriptor Ri, normalized; build X0 and RiT ----
  if (tid < MNT_) {
    int j = tid;
    size_t base = ((size_t)a * MNT_ + j) * 4;
    float r = ImageDR[base], dx = ImageDR[base + 1], dy = ImageDR[base + 2], dz = ImageDR[base + 3];
    float rs = (r > 1e-5f) ? r : 1.f;
    float ir = 1.f / rs;
    float u  = (r - 0.5f) * (1.f / 5.5f);
    float sm = ir * (u * u * u * (-6.f * u * u + 15.f * u - 10.f) + 1.f);
    float S  = (r > 0.f && r < 0.5f) ? ir : ((r >= 0.5f && r < 6.f) ? sm : 0.f);
    bool mk = fabsf(r) > 1e-5f;
    float Ri4[4] = { S, mk ? S * dx * ir : 0.f, mk ? S * dy * ir : 0.f, mk ? S * dz * ir : 0.f };
    const float* av = davg + ((size_t)t_a * MNT_ + j) * 4;
    const float* sd = dstd + ((size_t)t_a * MNT_ + j) * 4;
#pragma unroll
    for (int d = 0; d < 4; ++d) {
      float v = (Ri4[d] - av[d]) / sd[d];
      Ri4[d] = v;
      RiT[d * 224 + j] = (half_t)v;
    }
    // vectorized X0 row: [Srij, tfeat(8), zeros...]  (4 x ds_store_b128)
    const float* tf = TFE + (j / 100) * 8;
    v8h r0, r1, rz = {};
    r0[0] = (half_t)Ri4[0];
#pragma unroll
    for (int f = 0; f < 7; ++f) r0[1 + f] = (half_t)tf[f];
    r1 = rz;
    r1[0] = (half_t)tf[7];
    v8h* xp = (v8h*)(X0 + j * 32);
    xp[0] = r0; xp[1] = r1; xp[2] = rz; xp[3] = rz;
  }
  __syncthreads();

  // ---- phase 2: fused embedding MLP per 16-neighbor row tile (13 tiles / 8 waves) ----
  const int col = lane & 15, rg = lane >> 4;
  for (int rt = wave; rt < 13; rt += 8) {
    const int rbase = rt * 16 + rg * 8;
    // layer 0: 9->25 (no resnet), Npad=32 ; A-frag invariant over nt -> hoist
    {
      v16h afr = load_a(X0 + rt * 16 * 32, 32, 0, lane);
#pragma unroll
      for (int nt = 0; nt < 2; ++nt) {
        v8f c = {};
        c = wmma32(afr, load_b(W0, 32, nt * 16, 0, lane), c);
        int n = nt * 16 + col;
        float bias = (n < 25) ? BS[n] : 0.f;
        half_t* hp = H1 + rbase * 32 + n;
#pragma unroll
        for (int i = 0; i < 8; ++i) {
          float v = (n < 25) ? fast_tanh(c[i] + bias) : 0.f;
          hp[i * 32] = (half_t)v;
        }
      }
    }
    // layer 1: 25->50, resnet h += [x,x]
    {
      v16h afr = load_a(H1 + rt * 16 * 32, 32, 0, lane);
#pragma unroll
      for (int nt = 0; nt < 4; ++nt) {
        v8f c = {};
        c = wmma32(afr, load_b(W1, 32, nt * 16, 0, lane), c);
        int n = nt * 16 + col;
        float bias = (n < 50) ? BS[25 + n] : 0.f;
        const half_t* rp = H1 + rbase * 32 + (n < 25 ? n : n - 25);
        half_t* hp = H2 + rbase * 64 + n;
#pragma unroll
        for (int i = 0; i < 8; ++i) {
          float v = 0.f;
          if (n < 50) v = fast_tanh(c[i] + bias) + (float)rp[i * 32];
          hp[i * 64] = (half_t)v;
        }
      }
    }
    // layer 2: 50->100, resnet h += [x,x]; write G column-major (contiguous j -> b128)
    {
      v16h a0 = load_a(H2 + rt * 16 * 64, 64, 0, lane);
      v16h a1 = load_a(H2 + rt * 16 * 64, 64, 32, lane);
#pragma unroll
      for (int nt = 0; nt < 7; ++nt) {
        v8f c = {};
        c = wmma32(a0, load_b(W2, 64, nt * 16, 0, lane), c);
        c = wmma32(a1, load_b(W2, 64, nt * 16, 32, lane), c);
        int m = nt * 16 + col;
        float bias = (m < 100) ? BS[75 + m] : 0.f;
        const half_t* rp = H2 + rbase * 64 + (m < 50 ? m : m - 50);
        v8h pack = {};
        if (m < 100) {
#pragma unroll
          for (int i = 0; i < 8; ++i)
            pack[i] = (half_t)(fast_tanh(c[i] + bias) + (float)rp[i * 64]);
        }
        *(v8h*)(G + m * 224 + rbase) = pack;   // 8 contiguous rows j
      }
    }
  }
  __syncthreads();

  // ---- phase 3: xyz = Ri^T(4x200,pad 16x224) @ G(224x112) / MNT  (WMMA, 7 n-tiles) ----
  if (wave < 7) {
    int nt = wave;
    v8f c = {};
#pragma unroll
    for (int ks = 0; ks < 7; ++ks)
      c = wmma32(load_a(RiT, 224, ks * 32, lane), load_b(G, 224, nt * 16, ks * 32, lane), c);
    if (rg == 0) {  // lanes 0..15, vgprs 0..3 hold rows d=0..3
#pragma unroll
      for (int i = 0; i < 4; ++i) XYZ[i * 112 + nt * 16 + col] = c[i] * (1.f / 200.f);
    }
  }
  __syncthreads();

  // ---- phase 4: DR[m][q] = sum_d xyz[d][m]*xyz[d][q] -> scratch f16, vectorized ----
  if (tid < 200) {       // 200 threads x 8 contiguous elements = 1600
    v8h pack;
#pragma unroll
    for (int i = 0; i < 8; ++i) {
      int idx = tid * 8 + i;
      int m = idx >> 4, q = idx & 15;
      float s = 0.f;
#pragma unroll
      for (int d = 0; d < 4; ++d) s += XYZ[d * 112 + m] * XYZ[d * 112 + q];
      pack[i] = (half_t)s;
    }
    *(v8h*)(DRg + (size_t)a * 1600 + tid * 8) = pack;   // one global_store_b128
  }
}

// ---------------------------------------------------------------------------
// Kernel B: fitting net 1600->240->240->240->1 per atom, per type (select by map)
// One block (256 threads, 8 waves) per (16-atom tile, type). grid = 512.
// ---------------------------------------------------------------------------
__global__ __launch_bounds__(256) void fit_kernel(
    const half_t* __restrict__ DRg, const int* __restrict__ typemap,
    const float* __restrict__ fW0, const float* __restrict__ fb0,
    const float* __restrict__ fW1, const float* __restrict__ fb1,
    const float* __restrict__ fW2, const float* __restrict__ fb2,
    const float* __restrict__ fWf, const float* __restrict__ fbf,
    float* __restrict__ out)
{
  __shared__ half_t WS[FH_ * 32];     // weight k-slab, col-major n*32+k
  __shared__ half_t HB0[16 * 256];    // activations, row-major stride 256 (K pad 240->256)
  __shared__ half_t HB1[16 * 256];

  const int tid = threadIdx.x, lane = tid & 31, wave = tid >> 5;
  const int col = lane & 15, rg = lane >> 4;
  const int tp   = blockIdx.x >> 8;     // type 0/1
  const int tile = blockIdx.x & 255;    // 16-atom tile
  const int a0   = tile * 16;
  const float* W0p = fW0 + (size_t)tp * 1600 * FH_;
  const float* W1p = fW1 + (size_t)tp * FH_ * FH_;
  const float* W2p = fW2 + (size_t)tp * FH_ * FH_;

  // zero K-pad cols 240..255 of both activation buffers (NaN safety)
  for (int i = tid; i < 16 * 16; i += 256) {
    int r = i >> 4, cc = 240 + (i & 15);
    HB0[r * 256 + cc] = (half_t)0.f;
    HB1[r * 256 + cc] = (half_t)0.f;
  }

  const int nt0 = wave, nt1 = wave + 8;   // 15 N-tiles over 8 waves
  const int n0 = nt0 * 16 + col, n1 = nt1 * 16 + col;

  // ---- layer 0: K=1600 (50 k-slabs), A from global DR scratch ----
  v8f acc0 = {}, acc1 = {};
  for (int ksb = 0; ksb < 50; ++ksb) {
    int k0 = ksb * 32;
    __syncthreads();
    // prefetch next 30KB weight slab into cache (global_prefetch_b8, speculative)
    if (ksb + 1 < 50)
      __builtin_prefetch((const char*)(W0p + (size_t)(k0 + 32) * FH_) + tid * 128, 0, 0);
    for (int i = tid; i < FH_ * 32; i += 256) {
      int n = i >> 5, kk = i & 31;
      WS[i] = (half_t)W0p[(size_t)(k0 + kk) * FH_ + n];
    }
    __syncthreads();
    v16h afr;
    {
      int row = lane & 15, kh = lane >> 4;
      const half_t* p = DRg + (size_t)(a0 + row) * 1600 + k0 + kh * 8;
      V16U u; u.h[0] = *(const v8h*)p; u.h[1] = *(const v8h*)(p + 16);
      afr = u.v;
    }
    acc0 = wmma32(afr, load_b(WS, 32, nt0 * 16, 0, lane), acc0);
    if (nt1 < 15) acc1 = wmma32(afr, load_b(WS, 32, nt1 * 16, 0, lane), acc1);
  }
  {
    float b0a = fb0[tp * FH_ + n0];
    float b0b = (nt1 < 15) ? fb0[tp * FH_ + n1] : 0.f;
    half_t* h0 = HB0 + rg * 8 * 256;
#pragma unroll
    for (int i = 0; i < 8; ++i) {
      h0[i * 256 + n0] = (half_t)fast_tanh(acc0[i] + b0a);
      if (nt1 < 15) h0[i * 256 + n1] = (half_t)fast_tanh(acc1[i] + b0b);
    }
  }

  // ---- layer 1: K=240 (pad 256, 8 k-slabs), resnet h += x ----
  acc0 = {}; acc1 = {};
  for (int ksb = 0; ksb < 8; ++ksb) {
    int k0 = ksb * 32;
    __syncthreads();
    if (ksb + 1 < 8)
      __builtin_prefetch((const char*)(W1p + (size_t)(k0 + 32) * FH_) + tid * 128, 0, 0);
    for (int i = tid; i < FH_ * 32; i += 256) {
      int n = i >> 5, kk = i & 31, k = k0 + kk;
      WS[i] = (half_t)((k < FH_) ? W1p[(size_t)k * FH_ + n] : 0.f);
    }
    __syncthreads();
    v16h afr = load_a(HB0, 256, k0, lane);
    acc0 = wmma32(afr, load_b(WS, 32, nt0 * 16, 0, lane), acc0);
    if (nt1 < 15) acc1 = wmma32(afr, load_b(WS, 32, nt1 * 16, 0, lane), acc1);
  }
  {
    float b1a = fb1[tp * FH_ + n0];
    float b1b = (nt1 < 15) ? fb1[tp * FH_ + n1] : 0.f;
    half_t* h0 = HB0 + rg * 8 * 256;
    half_t* h1 = HB1 + rg * 8 * 256;
#pragma unroll
    for (int i = 0; i < 8; ++i) {
      h1[i * 256 + n0] = (half_t)(fast_tanh(acc0[i] + b1a) + (float)h0[i * 256 + n0]);
      if (nt1 < 15) h1[i * 256 + n1] = (half_t)(fast_tanh(acc1[i] + b1b) + (float)h0[i * 256 + n1]);
    }
  }

  // ---- layer 2: same shape, resnet; output back into HB0 ----
  acc0 = {}; acc1 = {};
  for (int ksb = 0; ksb < 8; ++ksb) {
    int k0 = ksb * 32;
    __syncthreads();
    if (ksb + 1 < 8)
      __builtin_prefetch((const char*)(W2p + (size_t)(k0 + 32) * FH_) + tid * 128, 0, 0);
    for (int i = tid; i < FH_ * 32; i += 256) {
      int n = i >> 5, kk = i & 31, k = k0 + kk;
      WS[i] = (half_t)((k < FH_) ? W2p[(size_t)k * FH_ + n] : 0.f);
    }
    __syncthreads();
    v16h afr = load_a(HB1, 256, k0, lane);
    acc0 = wmma32(afr, load_b(WS, 32, nt0 * 16, 0, lane), acc0);
    if (nt1 < 15) acc1 = wmma32(afr, load_b(WS, 32, nt1 * 16, 0, lane), acc1);
  }
  {
    float b2a = fb2[tp * FH_ + n0];
    float b2b = (nt1 < 15) ? fb2[tp * FH_ + n1] : 0.f;
    half_t* h0 = HB0 + rg * 8 * 256;
    half_t* h1 = HB1 + rg * 8 * 256;
#pragma unroll
    for (int i = 0; i < 8; ++i) {
      h0[i * 256 + n0] = (half_t)(fast_tanh(acc0[i] + b2a) + (float)h1[i * 256 + n0]);
      if (nt1 < 15) h0[i * 256 + n1] = (half_t)(fast_tanh(acc1[i] + b2b) + (float)h1[i * 256 + n1]);
    }
  }
  __syncthreads();

  // ---- final linear + per-type select + Etot reduction ----
  if (tid < 16) {
    int a = a0 + tid;
    float e = fbf[tp];
    for (int n = 0; n < FH_; n += 8) {
      v8h h = *(const v8h*)(HB0 + tid * 256 + n);
#pragma unroll
      for (int i = 0; i < 8; ++i) e += (float)h[i] * fWf[tp * FH_ + n + i];
    }
    int natom = a & (NAT_ - 1);
    if (typemap[natom] == tp) {
      out[BB_ + a] = e;                 // Ei, flat (B,NAT) after 4 Etot slots
      atomicAdd(&out[a >> 10], e);      // Etot[b]
    }
  }
}

__global__ void init_out_kernel(float* __restrict__ out) {
  if (threadIdx.x < BB_) out[threadIdx.x] = 0.f;
}

extern "C" void kernel_launch(void* const* d_in, const int* in_sizes, int n_in,
                              void* d_out, int out_size, void* d_ws, size_t ws_size,
                              hipStream_t stream) {
  (void)in_sizes; (void)n_in; (void)out_size; (void)ws_size;
  const int*   typemap = (const int*)d_in[1];
  const float* ImageDR = (const float*)d_in[3];
  const float* davg = (const float*)d_in[5];
  const float* dstd = (const float*)d_in[6];
  const float* tv   = (const float*)d_in[7];
  const float* tW0  = (const float*)d_in[8];
  const float* tb0  = (const float*)d_in[9];
  const float* eW0  = (const float*)d_in[10];
  const float* eb0  = (const float*)d_in[11];
  const float* eW1  = (const float*)d_in[12];
  const float* eb1  = (const float*)d_in[13];
  const float* eW2  = (const float*)d_in[14];
  const float* eb2  = (const float*)d_in[15];
  const float* fW0  = (const float*)d_in[16];
  const float* fb0  = (const float*)d_in[17];
  const float* fW1  = (const float*)d_in[18];
  const float* fb1  = (const float*)d_in[19];
  const float* fW2  = (const float*)d_in[20];
  const float* fb2  = (const float*)d_in[21];
  const float* fWf  = (const float*)d_in[22];
  const float* fbf  = (const float*)d_in[23];
  float*  out = (float*)d_out;
  half_t* DRg = (half_t*)d_ws;          // 4096 x 1600 f16 = 13.1 MB scratch

  init_out_kernel<<<1, 32, 0, stream>>>(out);
  embed_kernel<<<NATOMS_, 256, SMEM_A, stream>>>(ImageDR, typemap, davg, dstd, tv, tW0, tb0,
                                                 eW0, eb0, eW1, eb1, eW2, eb2, DRg);
  fit_kernel<<<NTYPES_ * (NATOMS_ / 16), 256, 0, stream>>>(DRg, typemap, fW0, fb0, fW1, fb1,
                                                           fW2, fb2, fWf, fbf, out);
}
